// SparseBraidRegister_41901700940049
// MI455X (gfx1250) — matrix operand order repaired and checked by reading, not compiled, not run on hardware
//
#include <hip/hip_runtime.h>
#include <hip/hip_bf16.h>
#include <cfloat>
#include <cstdint>
#include <cstddef>

#define B_   4
#define T_   2048
#define D_   2048
#define K_   7
#define DB_  128
#define TOPK_ 16
#define WIN_ 64

typedef __bf16 bf16_t;
typedef __attribute__((ext_vector_type(16))) __bf16 v16bf;
typedef __attribute__((ext_vector_type(8)))  float  v8f;

struct U4x2 { uint4 lo, hi; };  // 32 bytes == v16bf

static __device__ __forceinline__ v8f wmma_bf16(v16bf a, v16bf b, v8f c) {
  // D = A(16x32 bf16) * B(32x16 bf16) + C(16x16 f32)
  return __builtin_amdgcn_wmma_f32_16x16x32_bf16(false, a, false, b, (short)0, c,
                                                 false, false);
}

// Build a v16bf fragment from two 16B-aligned 128-bit loads (no element inserts)
static __device__ __forceinline__ v16bf load_frag2(const bf16_t* p0,
                                                   const bf16_t* p1) {
  U4x2 t;
  t.lo = *(const uint4*)p0;
  t.hi = *(const uint4*)p1;
  return __builtin_bit_cast(v16bf, t);
}

__device__ __forceinline__ float block_sum(float v, float* red) {
  const int tid = threadIdx.x;
  red[tid] = v;
  __syncthreads();
  for (int off = blockDim.x >> 1; off > 0; off >>= 1) {
    if (tid < off) red[tid] += red[tid + off];
    __syncthreads();
  }
  float r = red[0];
  __syncthreads();
  return r;
}

// ---------------------------------------------------------------------------
// init: fill salience buffer with -inf, zero the loss accumulators
// ---------------------------------------------------------------------------
__global__ void init_kernel(float* __restrict__ sal, float* Lp, float* Ld) {
  const size_t i = (size_t)blockIdx.x * blockDim.x + threadIdx.x;
  const size_t n = (size_t)B_ * T_ * WIN_;
  if (i < n) sal[i] = -FLT_MAX;
  if (i == 0) { *Lp = 0.f; *Ld = 0.f; }
}

// ---------------------------------------------------------------------------
// fp32 -> bf16 conversion (grid-stride)
// ---------------------------------------------------------------------------
__global__ void cvt_bf16_kernel(const float* __restrict__ x, bf16_t* __restrict__ y,
                                size_t n) {
  for (size_t i = (size_t)blockIdx.x * blockDim.x + threadIdx.x; i < n;
       i += (size_t)gridDim.x * blockDim.x)
    y[i] = (bf16_t)x[i];
}

// ---------------------------------------------------------------------------
// LayerNorm of X_write -> bf16 (one block per row)
// ---------------------------------------------------------------------------
__global__ __launch_bounds__(256)
void ln_write_kernel(const float* __restrict__ X, const float* __restrict__ g,
                     const float* __restrict__ bb, bf16_t* __restrict__ out) {
  __shared__ float red[256];
  const int row = blockIdx.x;
  const int tid = threadIdx.x;
  const float* x = X + (size_t)row * D_;
  float s = 0.f;
  for (int i = tid; i < D_; i += 256) s += x[i];
  const float mean = block_sum(s, red) * (1.0f / D_);
  float v = 0.f;
  for (int i = tid; i < D_; i += 256) { float d = x[i] - mean; v += d * d; }
  const float var = block_sum(v, red) * (1.0f / D_);
  const float rs = rsqrtf(var + 1e-5f);
  bf16_t* o = out + (size_t)row * D_;
  for (int i = tid; i < D_; i += 256)
    o[i] = (bf16_t)((x[i] - mean) * rs * g[i] + bb[i]);
}

// ---------------------------------------------------------------------------
// bf16 WMMA GEMM: C[M,N] = A[M,K] * B[N,K]^T
// block tile 128x64, 8 waves (4x2), each wave 32x32 via 2x2 wmma tiles, BK=32.
// DOUBLE-BUFFERED LDS: next k-tile's global loads are issued before the current
// tile's WMMAs, so global latency is hidden behind matrix math; one barrier
// per k-step. KD / NN / OUT_BF16 compile-time.
// ---------------------------------------------------------------------------
template <int KD, int NN, int OUT_BF16>
__global__ __launch_bounds__(256)
void gemm_bf16_wmma(const bf16_t* __restrict__ A, const bf16_t* __restrict__ Bw,
                    float* __restrict__ Cf, bf16_t* __restrict__ Cb) {
  __shared__ __align__(16) bf16_t As[2][128][32];
  __shared__ __align__(16) bf16_t Bs[2][64][32];
  const int tid  = threadIdx.x;
  const int lane = tid & 31;
  const int wv   = tid >> 5;
  const int wm   = wv & 3;
  const int wn   = wv >> 2;
  const int half = lane >> 4;
  const int l15  = lane & 15;
  const int bm = blockIdx.x * 128;
  const int bn = blockIdx.y * 64;

  v8f zero = {};
  v8f acc[2][2];
  for (int i = 0; i < 2; ++i)
    for (int j = 0; j < 2; ++j) acc[i][j] = zero;

  // per-thread staging addresses (invariant across k-steps)
  const int ar0 = tid >> 1, ac0 = (tid & 1) << 4;  // A: row tid>>1, cols [ac0,ac0+16)
  const int br0 = tid >> 2, bc0 = (tid & 3) << 3;  // B: row tid>>2, cols [bc0,bc0+8)
  const bf16_t* aSrc = A  + (size_t)(bm + ar0) * KD + ac0;
  const bf16_t* bSrc = Bw + (size_t)(bn + br0) * KD + bc0;

  // prologue: stage k-tile 0 into buffer 0
  {
    const uint4 ra0 = *(const uint4*)(aSrc);
    const uint4 ra1 = *(const uint4*)(aSrc + 8);
    const uint4 rb0 = *(const uint4*)(bSrc);
    *(uint4*)(&As[0][ar0][ac0])     = ra0;
    *(uint4*)(&As[0][ar0][ac0 + 8]) = ra1;
    *(uint4*)(&Bs[0][br0][bc0])     = rb0;
  }
  __syncthreads();

  for (int k0 = 0; k0 < KD; k0 += 32) {
    const int cur = (k0 >> 5) & 1;
    const int nxt = cur ^ 1;
    const bool more = (k0 + 32) < KD;

    // issue next tile's global loads first (latency overlapped by WMMAs below)
    uint4 na0 = {}, na1 = {}, nb0 = {};
    if (more) {
      na0 = *(const uint4*)(aSrc + k0 + 32);
      na1 = *(const uint4*)(aSrc + k0 + 40);
      nb0 = *(const uint4*)(bSrc + k0 + 32);
      if (k0 + 64 < KD)  // warm k-tile after next (global_prefetch_b8)
        __builtin_prefetch(aSrc + k0 + 64, 0, 1);
    }

    // compute on current buffer
    v16bf af[2], bfv[2];
    const int kb = half * 8;
#pragma unroll
    for (int mi = 0; mi < 2; ++mi) {
      const bf16_t* ap = &As[cur][wm * 32 + mi * 16 + l15][kb];
      af[mi] = load_frag2(ap, ap + 16);  // K {kb..kb+7} U {kb+16..kb+23}
    }
#pragma unroll
    for (int ni = 0; ni < 2; ++ni) {
      const bf16_t* bp = &Bs[cur][wn * 32 + ni * 16 + l15][kb];
      bfv[ni] = load_frag2(bp, bp + 16);  // K {kb..kb+7} U {kb+16..kb+23}
    }
#pragma unroll
    for (int mi = 0; mi < 2; ++mi)
#pragma unroll
      for (int ni = 0; ni < 2; ++ni)
        acc[mi][ni] = wmma_bf16(af[mi], bfv[ni], acc[mi][ni]);

    // commit next tile to the alternate buffer, then one barrier per k-step
    if (more) {
      *(uint4*)(&As[nxt][ar0][ac0])     = na0;
      *(uint4*)(&As[nxt][ar0][ac0 + 8]) = na1;
      *(uint4*)(&Bs[nxt][br0][bc0])     = nb0;
    }
    __syncthreads();
  }

#pragma unroll
  for (int mi = 0; mi < 2; ++mi)
#pragma unroll
    for (int ni = 0; ni < 2; ++ni)
#pragma unroll
      for (int r = 0; r < 8; ++r) {
        const int row = bm + wm * 32 + mi * 16 + half * 8 + r;
        const int col = bn + wn * 32 + ni * 16 + l15;
        const float v = acc[mi][ni][r];
        if (OUT_BF16) Cb[(size_t)row * NN + col] = (bf16_t)v;
        else          Cf[(size_t)row * NN + col] = v;
      }
}

// ---------------------------------------------------------------------------
// banded salience: sal[b, i*64+w] = dot(Xn[i], XW0[j]), j=i+w-64  (WMMA tiles)
// one wave per (b, i-tile, j-tile); 5 j-tiles cover [i0-64, i0+16)
// ---------------------------------------------------------------------------
__global__ __launch_bounds__(32)
void band_kernel(const bf16_t* __restrict__ Xn, const bf16_t* __restrict__ XW0,
                 float* __restrict__ sal) {
  const int nIt = T_ / 16;
  int id = blockIdx.x;
  const int b  = id / (nIt * 5);
  int rem      = id % (nIt * 5);
  const int it = rem / 5, jt = rem % 5;
  const int i0 = it * 16;
  const int j0 = i0 - 64 + jt * 16;
  if (j0 + 15 < 0) return;

  const int lane = threadIdx.x;
  const int half = lane >> 4, l15 = lane & 15;
  const int jr = j0 + l15;
  const int jc = jr < 0 ? 0 : jr;
  const bf16_t* arow = Xn  + ((size_t)b * T_ + i0 + l15) * D_ + half * 8;
  const bf16_t* brow = XW0 + ((size_t)b * T_ + jc) * D_ + half * 8;

  v8f acc = {};
  for (int k0 = 0; k0 < D_; k0 += 32) {
    const v16bf a  = load_frag2(arow + k0, arow + k0 + 16);
    const v16bf bv = load_frag2(brow + k0, brow + k0 + 16);
    acc = wmma_bf16(a, bv, acc);
  }
#pragma unroll
  for (int r = 0; r < 8; ++r) {
    const int i = i0 + half * 8 + r;
    const int j = j0 + l15;
    const int w = j - i + 64;
    if (j >= 0 && w >= 0 && w < 64)
      sal[((size_t)b * T_ + i) * WIN_ + w] = acc[r];
  }
}

// ---------------------------------------------------------------------------
// top-16 over T*64 scores per batch (one block per batch)
// ---------------------------------------------------------------------------
__global__ __launch_bounds__(256)
void topk_kernel(const float* __restrict__ sal, int* __restrict__ idx_out,
                 float* __restrict__ val_out) {
  const int b = blockIdx.x, tid = threadIdx.x;
  const float* s = sal + (size_t)b * T_ * WIN_;
  const int N = T_ * WIN_;
  float bv[TOPK_]; int bi[TOPK_];
#pragma unroll
  for (int i = 0; i < TOPK_; ++i) { bv[i] = -FLT_MAX; bi[i] = 0; }
  for (int i = tid; i < N; i += 256) {
    const float v = s[i];
    if (v > bv[TOPK_ - 1]) {
      int p = TOPK_ - 1;
      while (p > 0 && bv[p - 1] < v) { bv[p] = bv[p - 1]; bi[p] = bi[p - 1]; --p; }
      bv[p] = v; bi[p] = i;
    }
  }
  __shared__ float sv[256 * TOPK_];
  __shared__ int   si[256 * TOPK_];
  __shared__ float rv[256];
  __shared__ int   ri[256];
#pragma unroll
  for (int i = 0; i < TOPK_; ++i) { sv[tid * TOPK_ + i] = bv[i]; si[tid * TOPK_ + i] = bi[i]; }
  __syncthreads();
  for (int sel = 0; sel < TOPK_; ++sel) {
    float mv = -FLT_MAX; int mi = 0;
    for (int i = tid; i < 256 * TOPK_; i += 256)
      if (sv[i] > mv) { mv = sv[i]; mi = i; }
    rv[tid] = mv; ri[tid] = mi;
    __syncthreads();
    for (int off = 128; off > 0; off >>= 1) {
      if (tid < off && rv[tid + off] > rv[tid]) { rv[tid] = rv[tid + off]; ri[tid] = ri[tid + off]; }
      __syncthreads();
    }
    if (tid == 0) {
      const int wloc = ri[0];
      val_out[b * TOPK_ + sel] = rv[0];
      idx_out[b * TOPK_ + sel] = si[wloc];
      sv[wloc] = -FLT_MAX;
    }
    __syncthreads();
  }
}

// ---------------------------------------------------------------------------
// pair gather + relation:  R_t = (sum_p alpha_p (xn_i - xn_j)) @ Wv^T
// LN recomputed for the 32 selected rows; pair_sal == topk value (reused)
// ---------------------------------------------------------------------------
__global__ __launch_bounds__(256)
void pair_kernel(const float* __restrict__ X, const float* __restrict__ g,
                 const int* __restrict__ idx, const float* __restrict__ vals,
                 const float* __restrict__ Wv, float* __restrict__ Rt) {
  const int b = blockIdx.x, tid = threadIdx.x;
  __shared__ float u[D_];
  __shared__ float red[256];
  __shared__ float alpha[TOPK_];
  if (tid == 0) {
    float mx = -FLT_MAX;
    for (int p = 0; p < TOPK_; ++p) mx = fmaxf(mx, vals[b * TOPK_ + p]);
    float se = 0.f;
    for (int p = 0; p < TOPK_; ++p) { alpha[p] = expf(vals[b * TOPK_ + p] - mx); se += alpha[p]; }
    const float inv = 1.0f / se;
    for (int p = 0; p < TOPK_; ++p) alpha[p] *= inv;
  }
  for (int d = tid; d < D_; d += 256) u[d] = 0.f;
  __syncthreads();

  for (int p = 0; p < TOPK_; ++p) {
    const int id_ = idx[b * TOPK_ + p];
    const int i_ = id_ / WIN_;
    int j_ = i_ + (id_ % WIN_) - WIN_;
    if (j_ < 0) j_ = 0;
    const float* xi = X + ((size_t)b * T_ + i_) * D_;
    const float* xj = X + ((size_t)b * T_ + j_) * D_;
    float s = 0.f;
    for (int d = tid; d < D_; d += 256) s += xi[d];
    const float mi = block_sum(s, red) * (1.0f / D_);
    float v = 0.f;
    for (int d = tid; d < D_; d += 256) { float dd = xi[d] - mi; v += dd * dd; }
    const float rsi = rsqrtf(block_sum(v, red) * (1.0f / D_) + 1e-5f);
    s = 0.f;
    for (int d = tid; d < D_; d += 256) s += xj[d];
    const float mj = block_sum(s, red) * (1.0f / D_);
    v = 0.f;
    for (int d = tid; d < D_; d += 256) { float dd = xj[d] - mj; v += dd * dd; }
    const float rsj = rsqrtf(block_sum(v, red) * (1.0f / D_) + 1e-5f);
    const float a = alpha[p];
    for (int d = tid; d < D_; d += 256)
      u[d] += a * ((xi[d] - mi) * rsi - (xj[d] - mj) * rsj) * g[d];  // LN bias cancels
    __syncthreads();
  }
  if (tid < DB_) {
    float s = 0.f;
    const float* wrow = Wv + (size_t)tid * D_;
    for (int d = 0; d < D_; ++d) s += u[d] * wrow[d];
    Rt[b * DB_ + tid] = s;
  }
}

// ---------------------------------------------------------------------------
// slot write + strand mixer + aux losses (one block per batch; tiny matrices)
// ---------------------------------------------------------------------------
__global__ __launch_bounds__(256)
void slot_kernel(const float* __restrict__ slots_in, const float* __restrict__ Wk,
                 const float* __restrict__ skb, const float* __restrict__ temps,
                 const float* __restrict__ Wg_w, const float* __restrict__ Wg_b,
                 const float* __restrict__ ln_g, const float* __restrict__ ln_b,
                 const float* __restrict__ Wsq, const float* __restrict__ Wsk,
                 const float* __restrict__ Wsv, const float* __restrict__ lam,
                 const float* __restrict__ Rt, float* __restrict__ out_slots,
                 float* __restrict__ sn_ws, float* Lp, float* Ld) {
  const int b = blockIdx.x, tid = threadIdx.x;
  const float sd = 0.08838834764831845f;  // 1/sqrt(128)
  __shared__ float S[K_][DB_], UPD[K_][DB_], BLN[K_][DB_];
  __shared__ float Qm[K_][DB_], Km[K_][DB_], Vm[K_][DB_];
  __shared__ float Rr[DB_], swv[8], attm[K_][K_], lnm[K_], lnr[K_], red[256];

  for (int idx = tid; idx < K_ * DB_; idx += 256)
    S[idx >> 7][idx & 127] = slots_in[(size_t)(b * K_) * DB_ + idx];
  if (tid < DB_) Rr[tid] = Rt[b * DB_ + tid];
  __syncthreads();

  // slot_keys * R  (Qm reused as temp)
  for (int idx = tid; idx < K_ * DB_; idx += 256) {
    const int k = idx >> 7, e = idx & 127;
    float s = skb[k * DB_ + e];
    const float* wrow = Wk + (size_t)e * DB_;
    for (int f = 0; f < DB_; ++f) s += S[k][f] * wrow[f];
    Qm[k][e] = s * Rr[e];
  }
  __syncthreads();
  if (tid < K_) {
    float s = 0.f;
    for (int e = 0; e < DB_; ++e) s += Qm[tid][e];
    const float t = temps[tid];
    const float sp = (t > 20.f) ? t : log1pf(expf(t));
    swv[tid] = s * sd * sp;
  }
  __syncthreads();
  if (tid == 0) {
    float mx = -FLT_MAX;
    for (int k = 0; k < K_; ++k) mx = fmaxf(mx, swv[k]);
    float se = 0.f;
    for (int k = 0; k < K_; ++k) { swv[k] = expf(swv[k] - mx); se += swv[k]; }
    for (int k = 0; k < K_; ++k) swv[k] /= se;
  }
  __syncthreads();

  // gated write
  for (int idx = tid; idx < K_ * DB_; idx += 256) {
    const int k = idx >> 7, e = idx & 127;
    const float wr = swv[k] * Rr[e];
    float a = Wg_b[e];
    const float* wrow = Wg_w + (size_t)e * (2 * DB_);
    for (int f = 0; f < DB_; ++f)
      a += S[k][f] * wrow[f] + swv[k] * Rr[f] * wrow[DB_ + f];
    const float gg = 1.0f / (1.0f + expf(-a));
    UPD[k][e] = (1.0f - gg) * S[k][e] + gg * wr;
  }
  __syncthreads();

  // strand LN
  if (tid < K_) {
    float m = 0.f;
    for (int e = 0; e < DB_; ++e) m += UPD[tid][e];
    m *= (1.0f / DB_);
    float v = 0.f;
    for (int e = 0; e < DB_; ++e) { float d = UPD[tid][e] - m; v += d * d; }
    lnm[tid] = m;
    lnr[tid] = rsqrtf(v * (1.0f / DB_) + 1e-5f);
  }
  __syncthreads();
  for (int idx = tid; idx < K_ * DB_; idx += 256) {
    const int k = idx >> 7, e = idx & 127;
    BLN[k][e] = (UPD[k][e] - lnm[k]) * lnr[k] * ln_g[e] + ln_b[e];
  }
  __syncthreads();
  for (int idx = tid; idx < K_ * DB_; idx += 256) {
    const int k = idx >> 7, e = idx & 127;
    float q = 0.f, kk = 0.f, vv = 0.f;
    const float* wq = Wsq + (size_t)e * DB_;
    const float* wk = Wsk + (size_t)e * DB_;
    const float* wv = Wsv + (size_t)e * DB_;
    for (int f = 0; f < DB_; ++f) {
      const float x = BLN[k][f];
      q += x * wq[f]; kk += x * wk[f]; vv += x * wv[f];
    }
    Qm[k][e] = q; Km[k][e] = kk; Vm[k][e] = vv;
  }
  __syncthreads();
  if (tid < K_ * K_) {
    const int k = tid / K_, m = tid % K_;
    float s = 0.f;
    for (int e = 0; e < DB_; ++e) s += Qm[k][e] * Km[m][e];
    attm[k][m] = s * sd;
  }
  __syncthreads();
  if (tid < K_) {
    float mx = -FLT_MAX;
    for (int m = 0; m < K_; ++m) mx = fmaxf(mx, attm[tid][m]);
    float se = 0.f;
    for (int m = 0; m < K_; ++m) { attm[tid][m] = expf(attm[tid][m] - mx); se += attm[tid][m]; }
    for (int m = 0; m < K_; ++m) attm[tid][m] /= se;
  }
  __syncthreads();

  float lp = 0.f;
  for (int idx = tid; idx < K_ * DB_; idx += 256) {
    const int k = idx >> 7, e = idx & 127;
    float ctx = 0.f;
    for (int m = 0; m < K_; ++m) ctx += attm[k][m] * Vm[m][e];
    float l = tanhf(lam[e]);
    l = fminf(fmaxf(l, -1.0f), 1.0f);
    const float snv = UPD[k][e] + ctx * l;
    out_slots[(size_t)(b * K_) * DB_ + idx] = snv;
    sn_ws[(size_t)(b * K_) * DB_ + idx] = snv;
    const float d0 = snv - S[k][e];
    lp += d0 * d0;
    BLN[k][e] = snv;  // reuse for L_div
  }
  __syncthreads();
  const float tp = block_sum(lp, red);
  if (tid == 0) atomicAdd(Lp, tp / (float)(B_ * K_ * DB_));

  if (tid < K_) {
    float s = 0.f;
    for (int e = 0; e < DB_; ++e) s += BLN[tid][e] * BLN[tid][e];
    lnm[tid] = fmaxf(sqrtf(s), 1e-12f);
  }
  __syncthreads();
  if (tid < K_ * K_) {
    const int k = tid / K_, m = tid % K_;
    float c = 0.f;
    for (int e = 0; e < DB_; ++e) c += (BLN[k][e] / lnm[k]) * (BLN[m][e] / lnm[m]);
    attm[k][m] = (k == m) ? 0.f : c * c;
  }
  __syncthreads();
  if (tid == 0) {
    float s = 0.f;
    for (int k = 0; k < K_; ++k)
      for (int m = 0; m < K_; ++m) s += attm[k][m];
    atomicAdd(Ld, s / (float)(B_ * K_ * (K_ - 1)));
  }
}

// ---------------------------------------------------------------------------
// q_mean over T (column sums)
// ---------------------------------------------------------------------------
__global__ __launch_bounds__(256)
void qmean_kernel(const float* __restrict__ Xr, float* __restrict__ qm) {
  const int b = blockIdx.y;
  const int d = blockIdx.x * 256 + threadIdx.x;
  const float* p = Xr + (size_t)b * T_ * D_ + d;
  float s = 0.f;
  for (int t = 0; t < T_; ++t) s += p[(size_t)t * D_];
  qm[b * D_ + d] = s * (1.0f / T_);
}

// ---------------------------------------------------------------------------
// directional basis + render gate + k_r / gated v_r (one block per batch)
// ---------------------------------------------------------------------------
__global__ __launch_bounds__(256)
void gate_kernel(const float* __restrict__ qm, const float* __restrict__ W0,
                 const float* __restrict__ sn, const float* __restrict__ Wk_read,
                 const float* __restrict__ Wv_read, const float* __restrict__ rc,
                 float* __restrict__ kr, float* __restrict__ vg) {
  const int b = blockIdx.x, tid = threadIdx.x;
  __shared__ float q[D_], w0q[D_], red[256], basis[3], SN[K_][DB_];
  for (int d = tid; d < D_; d += 256) q[d] = qm[b * D_ + d];
  __syncthreads();
  for (int n = tid; n < D_; n += 256) {
    float s = 0.f;
    const float* wr = W0 + (size_t)n * D_;
    for (int d = 0; d < D_; ++d) s += q[d] * wr[d];
    w0q[n] = s;
  }
  __syncthreads();
  const float PHI = 1.618033988749895f, PSI = -0.6180339887498949f;
  const float idphi = 1.0f / (PHI - PSI);
  float p2 = 0.f, s2 = 0.f;
  for (int n = tid; n < D_; n += 256) {
    const float ph = (w0q[n] - PSI * q[n]) * idphi;
    const float ps = (PHI * q[n] - w0q[n]) * idphi;
    p2 += ph * ph; s2 += ps * ps;
  }
  p2 = block_sum(p2, red);
  s2 = block_sum(s2, red);
  if (tid == 0) {
    const float pm = sqrtf(p2), sm = sqrtf(s2);
    const float tot = pm + sm + 1e-6f;
    basis[0] = 1.0f; basis[1] = pm / tot; basis[2] = sm / tot;
  }
  for (int idx = tid; idx < K_ * DB_; idx += 256)
    SN[idx >> 7][idx & 127] = sn[(size_t)(b * K_) * DB_ + idx];
  __syncthreads();
  for (int idx = tid; idx < K_ * DB_; idx += 256) {
    const int k = idx >> 7, e = idx & 127;
    float s = 0.f;
    const float* wr = Wk_read + (size_t)e * DB_;
    for (int f = 0; f < DB_; ++f) s += SN[k][f] * wr[f];
    kr[(size_t)(b * K_) * DB_ + idx] = s;
  }
  for (int idx = tid; idx < K_ * D_; idx += 256) {
    const int k = idx / D_, d = idx % D_;
    float v = 0.f;
    const float* wr = Wv_read + (size_t)d * DB_;
    for (int e = 0; e < DB_; ++e) v += SN[k][e] * wr[e];
    const float gs = basis[0] * rc[((size_t)k * 3 + 0) * D_ + d] +
                     basis[1] * rc[((size_t)k * 3 + 1) * D_ + d] +
                     basis[2] * rc[((size_t)k * 3 + 2) * D_ + d];
    float gate = tanhf(gs);
    gate = fminf(fmaxf(gate, -1.0f), 1.0f);
    vg[(size_t)(b * K_) * D_ + idx] = v * gate;
  }
}

// ---------------------------------------------------------------------------
// read: softmax(q_r . k_r) @ (v_r*gate)  (one block per token row)
// ---------------------------------------------------------------------------
__global__ __launch_bounds__(128)
void z_kernel(const float* __restrict__ qr, const float* __restrict__ kr,
              const float* __restrict__ vg, float* __restrict__ z) {
  const int row = blockIdx.x;          // b*T + t
  const int b = row / T_;
  const int tid = threadIdx.x;
  const float sd = 0.08838834764831845f;
  __shared__ float red[128], p[8];
  const float q = qr[(size_t)row * DB_ + tid];
  for (int k = 0; k < K_; ++k) {
    red[tid] = q * kr[(size_t)(b * K_ + k) * DB_ + tid];
    __syncthreads();
    for (int off = 64; off > 0; off >>= 1) {
      if (tid < off) red[tid] += red[tid + off];
      __syncthreads();
    }
    if (tid == 0) p[k] = red[0] * sd;
    __syncthreads();
  }
  if (tid == 0) {
    float mx = -FLT_MAX;
    for (int k = 0; k < K_; ++k) mx = fmaxf(mx, p[k]);
    float se = 0.f;
    for (int k = 0; k < K_; ++k) { p[k] = expf(p[k] - mx); se += p[k]; }
    for (int k = 0; k < K_; ++k) p[k] /= se;
  }
  __syncthreads();
  for (int d = tid; d < D_; d += 128) {
    float s = 0.f;
    for (int k = 0; k < K_; ++k) s += p[k] * vg[(size_t)(b * K_ + k) * D_ + d];
    z[(size_t)row * D_ + d] = s;
  }
}

// ---------------------------------------------------------------------------
extern "C" void kernel_launch(void* const* d_in, const int* in_sizes, int n_in,
                              void* d_out, int out_size, void* d_ws, size_t ws_size,
                              hipStream_t stream) {
  (void)in_sizes; (void)n_in; (void)out_size; (void)ws_size;
  const float* X_write = (const float*)d_in[0];
  const float* X_read  = (const float*)d_in[1];
  const float* slots   = (const float*)d_in[2];
  const float* W0      = (const float*)d_in[3];
  const float* ln_w_g  = (const float*)d_in[4];
  const float* ln_w_b  = (const float*)d_in[5];
  const float* Wv      = (const float*)d_in[6];
  const float* Wk      = (const float*)d_in[7];
  const float* skb     = (const float*)d_in[8];
  const float* Wg_w    = (const float*)d_in[9];
  const float* Wg_b    = (const float*)d_in[10];
  const float* ln_s_g  = (const float*)d_in[11];
  const float* ln_s_b  = (const float*)d_in[12];
  const float* Wsq     = (const float*)d_in[13];
  const float* Wsk     = (const float*)d_in[14];
  const float* Wsv     = (const float*)d_in[15];
  const float* lam     = (const float*)d_in[16];
  const float* Wq_read = (const float*)d_in[17];
  const float* Wk_read = (const float*)d_in[18];
  const float* Wv_read = (const float*)d_in[19];
  const float* rc      = (const float*)d_in[20];
  const float* temps   = (const float*)d_in[21];

  float* out = (float*)d_out;
  const size_t ZN = (size_t)B_ * T_ * D_;
  float* out_slots = out + ZN;
  float* Lp = out + ZN + (size_t)B_ * K_ * DB_;
  float* Ld = Lp + 1;

  uint8_t* w = (uint8_t*)d_ws;
  auto take = [&](size_t bytes) -> void* {
    void* p = (void*)w;
    w += (bytes + 255) & ~(size_t)255;
    return p;
  };
  bf16_t* Xn_bf  = (bf16_t*)take((size_t)B_ * T_ * D_ * 2);
  bf16_t* XW0_bf = (bf16_t*)take((size_t)B_ * T_ * D_ * 2);
  bf16_t* Xr_bf  = (bf16_t*)take((size_t)B_ * T_ * D_ * 2);
  bf16_t* W0_bf  = (bf16_t*)take((size_t)D_ * D_ * 2);
  bf16_t* Wq_bf  = (bf16_t*)take((size_t)DB_ * D_ * 2);
  float*  sal    = (float*)take((size_t)B_ * T_ * WIN_ * 4);
  float*  qr     = (float*)take((size_t)B_ * T_ * DB_ * 4);
  int*    tidx   = (int*)take((size_t)B_ * TOPK_ * 4);
  float*  tval   = (float*)take((size_t)B_ * TOPK_ * 4);
  float*  Rt     = (float*)take((size_t)B_ * DB_ * 4);
  float*  sn_ws  = (float*)take((size_t)B_ * K_ * DB_ * 4);
  float*  kr     = (float*)take((size_t)B_ * K_ * DB_ * 4);
  float*  vg     = (float*)take((size_t)B_ * K_ * D_ * 4);
  float*  qm     = (float*)take((size_t)B_ * D_ * 4);

  init_kernel<<<2048, 256, 0, stream>>>(sal, Lp, Ld);
  cvt_bf16_kernel<<<2048, 256, 0, stream>>>(W0, W0_bf, (size_t)D_ * D_);
  cvt_bf16_kernel<<<256, 256, 0, stream>>>(Wq_read, Wq_bf, (size_t)DB_ * D_);
  cvt_bf16_kernel<<<4096, 256, 0, stream>>>(X_read, Xr_bf, (size_t)B_ * T_ * D_);
  ln_write_kernel<<<B_ * T_, 256, 0, stream>>>(X_write, ln_w_g, ln_w_b, Xn_bf);

  // X_W0 = Xn @ W0^T  (M=8192, N=2048, K=2048), bf16 out for band WMMA
  gemm_bf16_wmma<D_, D_, 1><<<dim3(64, 32), 256, 0, stream>>>(Xn_bf, W0_bf,
                                                              nullptr, XW0_bf);
  band_kernel<<<B_ * (T_ / 16) * 5, 32, 0, stream>>>(Xn_bf, XW0_bf, sal);
  topk_kernel<<<B_, 256, 0, stream>>>(sal, tidx, tval);
  pair_kernel<<<B_, 256, 0, stream>>>(X_write, ln_w_g, tidx, tval, Wv, Rt);
  slot_kernel<<<B_, 256, 0, stream>>>(slots, Wk, skb, temps, Wg_w, Wg_b, ln_s_g,
                                      ln_s_b, Wsq, Wsk, Wsv, lam, Rt, out_slots,
                                      sn_ws, Lp, Ld);
  // q_r = X_read @ Wq_read^T  (M=8192, N=128, K=2048), fp32 out
  gemm_bf16_wmma<D_, DB_, 0><<<dim3(64, 2), 256, 0, stream>>>(Xr_bf, Wq_bf,
                                                              qr, nullptr);
  qmean_kernel<<<dim3(D_ / 256, B_), 256, 0, stream>>>(X_read, qm);
  gate_kernel<<<B_, 256, 0, stream>>>(qm, W0, sn_ws, Wk_read, Wv_read, rc, kr, vg);
  z_kernel<<<B_ * T_, 128, 0, stream>>>(qr, kr, vg, out);
}
